// CenterGeneration_21672404976127
// MI455X (gfx1250) — matrix operand order repaired and checked by reading, not compiled, not run on hardware
//
#include <hip/hip_runtime.h>

typedef __attribute__((ext_vector_type(16))) __bf16 v16bf;
typedef __attribute__((ext_vector_type(8)))  float  v8f;
typedef int v4i __attribute__((vector_size(16)));
typedef unsigned int   u32;
typedef unsigned short u16;

struct alignas(16) B128 { u32 x[4]; };

union Frag16 {
  v16bf v;
  u16   u[16];
  B128  q[2];
};

#define B_      16
#define CIN     256
#define HH      128
#define WW      128
#define HCONV   64
#define NHM     80
#define KTOT    2304          // 256 * 9
#define HW      16384         // 128*128
#define TOPK    300
#define HM_ELEMS (B_*NHM*HW)  // 20971520

// A-stage: 3 rows x 132 cols x 32 ch (bf16), channel-innermost
#define A_ELEMS (3*132*32)    // 12672 u16 -> 25344 B
// B-stage: 64 n x 9 taps x 32 ch (bf16) viewed as u32 pairs
#define B_U32   (HCONV*9*16)  // 9216 u32 -> 36864 B
#define BUF_BYTES (A_ELEMS*2 + B_U32*4)   // 62208

#if defined(__has_builtin)
# if __has_builtin(__builtin_amdgcn_global_load_async_to_lds_b128)
#  define ASYNC_OK 1
# endif
#endif
#ifndef ASYNC_OK
# define ASYNC_OK 0
#endif

__device__ __forceinline__ u16 f2bf(float f) {
  u32 u = __float_as_uint(f);
  u32 r = u + 0x7FFFu + ((u >> 16) & 1u);   // round-to-nearest-even
  return (u16)(r >> 16);
}

#if ASYNC_OK
__device__ __forceinline__ void wait_async0() {
# if __has_builtin(__builtin_amdgcn_s_wait_asynccnt)
  __builtin_amdgcn_s_wait_asynccnt(0);
# else
  asm volatile("s_wait_asynccnt 0x0" ::: "memory");
# endif
}
#endif

// ---------------------------------------------------------------------------
// Kernel 0: repack conv weights fp32 -> bf16.
// w1t layout [n][tap][c]  (tap-major K ordering, c contiguous)
// w2t layout [n][c]       (identity)
// ---------------------------------------------------------------------------
__global__ void prep_weights(const float* __restrict__ w1, const float* __restrict__ w2,
                             u16* __restrict__ w1t, u16* __restrict__ w2t) {
  int i = blockIdx.x * blockDim.x + threadIdx.x;
  if (i < HCONV * KTOT) {
    const int n   = i / KTOT;
    const int r   = i - n * KTOT;
    const int tap = r >> 8;           // 0..8
    const int c   = r & 255;
    w1t[i] = f2bf(w1[(n * CIN + c) * 9 + tap]);
  } else {
    int j = i - HCONV * KTOT;
    if (j < NHM * HCONV) w2t[j] = f2bf(w2[j]);
  }
}

// ---------------------------------------------------------------------------
// Kernel 1: 3x3 conv 256->64 + bias + ReLU, implicit GEMM, bf16 WMMA,
// f32 accumulate.  One workgroup = one (b, y) image row.  M=128, N=64,
// K=2304 processed as 8 channel-chunks x 9 taps (K=32 slices).
// Double-buffered LDS staging; weights DMA'd via async global->LDS b128
// (ASYNCcnt) when available, activations converted f32->bf16 in flight.
// Fragment loads software-pipelined so WMMAs never wait on dscnt==0.
// ---------------------------------------------------------------------------
__global__ __launch_bounds__(256) void conv1_wmma(
    const float* __restrict__ feat, const u32* __restrict__ w1t32,
    const float* __restrict__ b1, u16* __restrict__ hws) {
  extern __shared__ char dynsmem[];
  u16* aL0 = (u16*)(dynsmem);
  u32* bL0 = (u32*)(dynsmem + A_ELEMS * 2);
  u16* aL1 = (u16*)(dynsmem + BUF_BYTES);
  u32* bL1 = (u32*)(dynsmem + BUF_BYTES + A_ELEMS * 2);

  const int tid  = threadIdx.x;
  const int lane = tid & 31;
  const int wave = tid >> 5;
  const int g    = lane >> 4;
  const int ln   = lane & 15;
  const int b    = blockIdx.x >> 7;
  const int y    = blockIdx.x & 127;
  const int x0   = wave << 4;
  const int xm   = x0 + ln;

  v8f zero = {0.f, 0.f, 0.f, 0.f, 0.f, 0.f, 0.f, 0.f};
  v8f acc[4];
#pragma unroll
  for (int i = 0; i < 4; ++i) acc[i] = zero;

  float areg[50];
#if !ASYNC_OK
  u32 breg[36];
#endif

  auto loadA_regs = [&](int c0) {
#pragma unroll
    for (int j = 0; j < 50; ++j) {
      const int idx = tid + (j << 8);
      float v = 0.f;
      if (idx < A_ELEMS) {
        const int cl  = idx / 396;            // channel 0..31
        const int rem = idx - cl * 396;       // r3*132 + xo
        const int r3  = rem / 132;
        const int xo  = rem - r3 * 132;
        const int gx  = xo - 1;
        const int gy  = y + r3 - 1;
        if ((unsigned)gx < (unsigned)WW && (unsigned)gy < (unsigned)HH)
          v = feat[(((b * CIN + c0 + cl) * HH + gy) << 7) + gx];
      }
      areg[j] = v;
    }
  };

  auto storeA_lds = [&](u16* aL) {
#pragma unroll
    for (int j = 0; j < 50; ++j) {
      const int idx = tid + (j << 8);
      if (idx < A_ELEMS) {
        const int cl  = idx / 396;
        const int rem = idx - cl * 396;       // r3*132 + xo
        aL[rem * 32 + cl] = f2bf(areg[j]);    // channel-innermost
      }
    }
  };

#if ASYNC_OK
  // DMA bf16 weights straight into LDS: 2304 16-byte units per chunk.
  auto issueB_async = [&](int c0, u32* bL) {
#pragma unroll
    for (int t = 0; t < 9; ++t) {
      const int u = tid + (t << 8);           // 16B-unit index, < 2304
      const char* gsrc = (const char*)w1t32 + (u >> 2) * 512 + c0 * 2 + (u & 3) * 16;
      char* ldst = (char*)bL + u * 16;
      __builtin_amdgcn_global_load_async_to_lds_b128(
          (v4i*)const_cast<char*>(gsrc), (v4i*)ldst, 0, 0);
    }
  };
#else
  auto loadB_regs = [&](int c0) {
#pragma unroll
    for (int t = 0; t < 36; ++t) {
      const int j = tid + (t << 8);           // [ntap][cpair], 9216 total
      breg[t] = w1t32[((j >> 4) << 7) + (c0 >> 1) + (j & 15)];
    }
  };
  auto storeB_lds = [&](u32* bL) {
#pragma unroll
    for (int t = 0; t < 36; ++t) bL[tid + (t << 8)] = breg[t];
  };
#endif

  auto loadA = [&](Frag16& f, const u16* aL, int tap) {
    const int dy = tap / 3, dx = tap % 3;
    const u16* ap = aL + ((dy * 132) + (xm + dx)) * 32;
    f.q[0] = *reinterpret_cast<const B128*>(ap + 16 * g);
    f.q[1] = *reinterpret_cast<const B128*>(ap + 16 * g + 8);
  };
  auto loadB = [&](Frag16& f, const u32* bL, int tap, int nt) {
    const int n = nt * 16 + ln;
    const u16* bp = reinterpret_cast<const u16*>(bL) + (n * 9 + tap) * 32 + 16 * g;
    f.q[0] = *reinterpret_cast<const B128*>(bp);
    f.q[1] = *reinterpret_cast<const B128*>(bp + 8);
  };

  // 36-step flattened tap x nt loop with pipelined fragment loads
  auto compute = [&](const u16* aL, const u32* bL) {
    Frag16 a[2], bm[2];
    loadA(a[0], aL, 0);
    loadB(bm[0], bL, 0, 0);
#pragma unroll
    for (int i = 0; i < 36; ++i) {
      const int tap = i >> 2, nt = i & 3;
      const int ni = i + 1;
      if (ni < 36) {
        const int ntap = ni >> 2, nnt = ni & 3;
        if (nnt == 0) loadA(a[ntap & 1], aL, ntap);
        loadB(bm[ni & 1], bL, ntap, nnt);
      }
      acc[nt] = __builtin_amdgcn_wmma_f32_16x16x32_bf16(
          false, a[tap & 1].v, false, bm[i & 1].v, (short)0, acc[nt], false, false);
    }
  };

  // prologue: stage chunk 0
  loadA_regs(0);
  storeA_lds(aL0);
#if ASYNC_OK
  issueB_async(0, bL0);
#else
  loadB_regs(0);
  storeB_lds(bL0);
#endif

  for (int chunk = 0; chunk < 8; ++chunk) {
#if ASYNC_OK
    wait_async0();                            // our async B stores landed
#endif
    __syncthreads();                          // publish staged buffer
    const bool last = (chunk == 7);
    const u16* aL = (chunk & 1) ? aL1 : aL0;
    const u32* bL = (chunk & 1) ? bL1 : bL0;
    u16* aN = (chunk & 1) ? aL0 : aL1;
    u32* bN = (chunk & 1) ? bL0 : bL1;
    if (!last) {
      loadA_regs((chunk + 1) * 32);           // overlap with WMMAs below
#if ASYNC_OK
      issueB_async((chunk + 1) * 32, bN);
#else
      loadB_regs((chunk + 1) * 32);
#endif
    }
    compute(aL, bL);
    if (!last) {
      storeA_lds(aN);
#if !ASYNC_OK
      storeB_lds(bN);
#endif
    }
  }

  // epilogue: bias + ReLU, store bf16 NHWC ([b][y][x][64])
#pragma unroll
  for (int nt = 0; nt < 4; ++nt) {
    const int ch    = nt * 16 + ln;
    const float bia = b1[ch];
#pragma unroll
    for (int r = 0; r < 8; ++r) {
      float v = acc[nt][r] + bia;
      v = v > 0.f ? v : 0.f;
      const int x = x0 + r + 8 * g;
      hws[(((b << 7) + y) * 128 + x) * HCONV + ch] = f2bf(v);
    }
  }
}

// ---------------------------------------------------------------------------
// Kernel 2: 1x1 conv 64->80 + bias + sigmoid + mask, bf16 WMMA.
// M = B*H*W, N = 80, K = 64.  NHWC-bf16 input -> contiguous A loads.
// ---------------------------------------------------------------------------
__global__ __launch_bounds__(256) void conv2_wmma(
    const u16* __restrict__ hws, const u16* __restrict__ w2t,
    const float* __restrict__ b2, const unsigned char* __restrict__ mask,
    float* __restrict__ out) {
  const int lane = threadIdx.x & 31;
  const int wave = threadIdx.x >> 5;
  const int g    = lane >> 4;
  const int ln   = lane & 15;
  const int mt   = blockIdx.x * 8 + wave;
  const int m0   = mt << 4;

  v8f zero = {0.f, 0.f, 0.f, 0.f, 0.f, 0.f, 0.f, 0.f};
  v8f acc[5];
#pragma unroll
  for (int i = 0; i < 5; ++i) acc[i] = zero;

  const u16* arow = hws + (m0 + ln) * HCONV;
#pragma unroll
  for (int s = 0; s < 2; ++s) {
    Frag16 a;
    a.q[0] = *reinterpret_cast<const B128*>(arow + s * 32 + 16 * g);
    a.q[1] = *reinterpret_cast<const B128*>(arow + s * 32 + 16 * g + 8);
#pragma unroll
    for (int nt = 0; nt < 5; ++nt) {
      Frag16 bm;
      const B128* bp = reinterpret_cast<const B128*>(
          w2t + (nt * 16 + ln) * HCONV + s * 32 + g * 16);
      bm.q[0] = bp[0];
      bm.q[1] = bp[1];
      acc[nt] = __builtin_amdgcn_wmma_f32_16x16x32_bf16(
          false, a.v, false, bm.v, (short)0, acc[nt], false, false);
    }
  }

#pragma unroll
  for (int nt = 0; nt < 5; ++nt) {
    const int n     = nt * 16 + ln;
    const float bia = b2[n];
#pragma unroll
    for (int r = 0; r < 8; ++r) {
      const int m   = m0 + r + 8 * g;
      const int bb  = m >> 14;
      const int rem = m & 16383;
      float v = acc[nt][r] + bia;
      v = 1.f / (1.f + __expf(-v));
      if (mask[bb * HW + rem]) v = 0.f;
      out[(bb * NHM + n) * HW + rem] = v;
    }
  }
}

// ---------------------------------------------------------------------------
// wave32 argmax reduction (value desc, index asc tie-break)
// ---------------------------------------------------------------------------
__device__ __forceinline__ void waveArgmax(float& bv, int& bi) {
#pragma unroll
  for (int off = 16; off > 0; off >>= 1) {
    const float ov = __shfl_down(bv, off, 32);
    const int   oi = __shfl_down(bi, off, 32);
    if (ov > bv || (ov == bv && oi < bi)) { bv = ov; bi = oi; }
  }
}

// ---------------------------------------------------------------------------
// Kernel 3: per-(b,cat) 3x3 NMS + top-300.  plane + heat both in LDS;
// shuffle-based argmax, 3 barriers per extraction.
// ---------------------------------------------------------------------------
__global__ __launch_bounds__(256) void nms_topk1(const float* __restrict__ hm,
                                                 float* __restrict__ s1s,
                                                 int* __restrict__ s1p) {
  extern __shared__ char dynsmem[];
  float* smem  = (float*)dynsmem;
  float* plane = smem;                          // 16384
  float* heat  = smem + HW;                     // 16384
  float* redv  = smem + 2 * HW;                 // 8 (padded to 16)
  int*   redi  = (int*)(smem + 2 * HW + 16);    // 8

  const int tid = threadIdx.x;
  const int lane = tid & 31;
  const int wave = tid >> 5;
  const int cat = blockIdx.x;
  const int b   = blockIdx.y;
  const float* src = hm + (b * NHM + cat) * HW;

  for (int i = tid; i < HW; i += 256) plane[i] = src[i];
  __syncthreads();

  for (int j = 0; j < 64; ++j) {
    const int idx = tid + (j << 8);
    const int y = idx >> 7, x = idx & 127;
    const float c = plane[idx];
    float mx = c;
    for (int dy = -1; dy <= 1; ++dy)
      for (int dx = -1; dx <= 1; ++dx) {
        const int yy = y + dy, xx = x + dx;
        if ((unsigned)yy < 128u && (unsigned)xx < 128u) {
          const float nb = plane[(yy << 7) + xx];
          mx = nb > mx ? nb : mx;
        }
      }
    heat[idx] = (c == mx) ? c : 0.f;
  }
  __syncthreads();

  const int obase = (b * NHM + cat) * TOPK;
  for (int it = 0; it < TOPK; ++it) {
    float bv = -3.0e38f;
    int bi = 0;
    for (int j = 0; j < 64; ++j) {
      const int idx = tid + (j << 8);
      const float v = heat[idx];
      if (v > bv) { bv = v; bi = idx; }
    }
    waveArgmax(bv, bi);
    if (lane == 0) { redv[wave] = bv; redi[wave] = bi; }
    __syncthreads();
    if (tid == 0) {
      float wv = redv[0]; int wi = redi[0];
      for (int w = 1; w < 8; ++w) {
        const float ov = redv[w]; const int oi = redi[w];
        if (ov > wv || (ov == wv && oi < wi)) { wv = ov; wi = oi; }
      }
      redv[0] = wv; redi[0] = wi;
      s1s[obase + it] = wv;
      s1p[obase + it] = wi;
    }
    __syncthreads();
    const int wi = redi[0];
    if (tid == (wi & 255)) heat[wi] = -3.0e38f;
    __syncthreads();
  }
}

// ---------------------------------------------------------------------------
// Kernel 4: per-batch top-300 over 80*300 candidates + refpoint normalize.
// ---------------------------------------------------------------------------
__global__ __launch_bounds__(256) void topk2(const float* __restrict__ s1s,
                                             const int* __restrict__ s1p,
                                             const float* __restrict__ vr,
                                             float* __restrict__ rp) {
  extern __shared__ char dynsmem[];
  float* smem = (float*)dynsmem;
  const int NC = NHM * TOPK;                    // 24000
  float* sc   = smem;
  float* redv = smem + NC;                      // 8 (padded 16)
  int*   redi = (int*)(smem + NC + 16);

  const int tid = threadIdx.x;
  const int lane = tid & 31;
  const int wave = tid >> 5;
  const int b   = blockIdx.x;

  for (int i = tid; i < NC; i += 256) sc[i] = s1s[b * NC + i];
  __syncthreads();

  const float dx = 128.f * vr[b * 2 + 1];
  const float dy = 128.f * vr[b * 2 + 0];

  for (int it = 0; it < TOPK; ++it) {
    float bv = -3.0e38f;
    int bi = 0;
    for (int i = tid; i < NC; i += 256) {
      const float v = sc[i];
      if (v > bv) { bv = v; bi = i; }
    }
    waveArgmax(bv, bi);
    if (lane == 0) { redv[wave] = bv; redi[wave] = bi; }
    __syncthreads();
    if (tid == 0) {
      float wv = redv[0]; int wi = redi[0];
      for (int w = 1; w < 8; ++w) {
        const float ov = redv[w]; const int oi = redi[w];
        if (ov > wv || (ov == wv && oi < wi)) { wv = ov; wi = oi; }
      }
      redi[0] = wi;
      const int pos  = s1p[b * NC + wi];
      const float xs = (float)(pos & 127) + 0.5f;
      const float ys = (float)(pos >> 7) + 0.5f;
      rp[(b * TOPK + it) * 2 + 0] = xs / dx;
      rp[(b * TOPK + it) * 2 + 1] = ys / dy;
    }
    __syncthreads();
    const int wi = redi[0];
    if (tid == (wi & 255)) sc[wi] = -3.0e38f;
    __syncthreads();
  }
}

// ---------------------------------------------------------------------------
extern "C" void kernel_launch(void* const* d_in, const int* in_sizes, int n_in,
                              void* d_out, int out_size, void* d_ws, size_t ws_size,
                              hipStream_t stream) {
  (void)in_sizes; (void)n_in; (void)out_size; (void)ws_size;
  const float*         feat = (const float*)d_in[0];
  const unsigned char* mask = (const unsigned char*)d_in[1];
  const float*         vr   = (const float*)d_in[2];
  const float*         w1   = (const float*)d_in[6];
  const float*         b1   = (const float*)d_in[7];
  const float*         w2   = (const float*)d_in[8];
  const float*         b2   = (const float*)d_in[9];
  float* out = (float*)d_out;

  char* ws = (char*)d_ws;
  u16*   w1t = (u16*)(ws);                                      //   294,912 B
  u16*   w2t = (u16*)(ws + 294912);                             //    10,240 B
  u16*   hws = (u16*)(ws + 305152);                             // 33,554,432 B
  float* s1s = (float*)(ws + 305152 + 33554432);                //  1,536,000 B
  int*   s1p = (int*)  (ws + 305152 + 33554432 + 1536000);      //  1,536,000 B

  prep_weights<<<596, 256, 0, stream>>>(w1, w2, w1t, w2t);
  conv1_wmma<<<2048, 256, 2 * BUF_BYTES, stream>>>(feat, (const u32*)w1t, b1, hws);
  conv2_wmma<<<2048, 256, 0, stream>>>(hws, w2t, b2, mask, out);
  nms_topk1<<<dim3(80, 16), dim3(256), (2 * HW + 32) * sizeof(float), stream>>>(out, s1s, s1p);
  topk2<<<16, 256, (NHM * TOPK + 32) * sizeof(float), stream>>>(s1s, s1p, vr, out + HM_ELEMS);
}
